// EquivariantBlock_19911468384607
// MI455X (gfx1250) — compile-verified, matrix operand order checked by reading, not run on hardware
//
#include <hip/hip_runtime.h>
#include <hip/hip_bf16.h>
#include <math.h>

typedef __bf16 bf16_t;
typedef bf16_t v16bf __attribute__((ext_vector_type(16)));
typedef bf16_t v8bf  __attribute__((ext_vector_type(8)));
typedef float  v8f   __attribute__((ext_vector_type(8)));

#define FDIM 64
#define KDIM 128
#define MIN_DIM 384   // 2F + (F + K) + F
#define NGRAPH 64
#define NFRAG 80      // 48 (Wm1) + 8 (Wm2) + 8 (Wx1) + 16 (Wx2)

__device__ __forceinline__ v8f wmma_bf16(v16bf a, v16bf b, v8f c) {
  return __builtin_amdgcn_wmma_f32_16x16x32_bf16(false, a, false, b, (short)0, c,
                                                 false, false);
}

// Pre-packed fragment: Wf[frag*512 + lane*16 + idx]; one 32B per-lane load.
__device__ __forceinline__ v16bf load_frag(const bf16_t* __restrict__ Wf, int frag,
                                           int lane) {
  return *(const v16bf*)(Wf + ((size_t)frag << 9) + (lane << 4));
}

__device__ __forceinline__ v8bf cvt8(float4 a, float4 b) {
  v8bf r;
  r[0] = (bf16_t)a.x; r[1] = (bf16_t)a.y; r[2] = (bf16_t)a.z; r[3] = (bf16_t)a.w;
  r[4] = (bf16_t)b.x; r[5] = (bf16_t)b.y; r[6] = (bf16_t)b.z; r[7] = (bf16_t)b.w;
  return r;
}

// ---------------- preprocessing kernels ----------------

__global__ void zero_kernel(float* p, int n) {
  int i = blockIdx.x * blockDim.x + threadIdx.x;
  if (i < n) p[i] = 0.f;
}

// Pack the four weight matrices into WMMA B-fragment-major bf16 layout.
// Fragment f: 32 lanes x 16 bf16.  lane<16 -> K {kb..kb+7, kb+16..kb+23},
// lane>=16 -> K {kb+8..kb+15, kb+24..kb+31}; n = t*16 + (lane&15).
__global__ void wpack_kernel(const float* __restrict__ Wm1, const float* __restrict__ Wm2,
                             const float* __restrict__ Wx1, const float* __restrict__ Wx2,
                             bf16_t* __restrict__ Wf) {
  int i = blockIdx.x * blockDim.x + threadIdx.x;
  if (i >= NFRAG * 512) return;
  int f = i >> 9;
  int within = i & 511;
  int lane = within >> 4;
  int idx = within & 15;
  int g = idx >> 3, j = idx & 7;
  int half = lane >> 4, nl = lane & 15;
  const float* W;
  int Ndim, kb, t;
  if (f < 48)      { W = Wm1; Ndim = FDIM; kb = (f >> 2) * 32; t = f & 3; }
  else if (f < 56) { int ff = f - 48; W = Wm2; Ndim = FDIM; kb = (ff >> 2) * 32; t = ff & 3; }
  else if (f < 64) { int ff = f - 56; W = Wx1; Ndim = FDIM; kb = (ff >> 2) * 32; t = ff & 3; }
  else             { int ff = f - 64; W = Wx2; Ndim = KDIM; kb = (ff >> 3) * 32; t = ff & 7; }
  int k = kb + g * 16 + half * 8 + j;
  int n = t * 16 + nl;
  Wf[i] = (bf16_t)W[(size_t)k * Ndim + n];
}

__global__ void accumx_kernel(const int* __restrict__ batch, const float* __restrict__ X,
                              float* __restrict__ sumX, float* __restrict__ cnt, int N) {
  int i = blockIdx.x * blockDim.x + threadIdx.x;
  if (i >= N * KDIM) return;
  int n = i >> 7, k = i & (KDIM - 1);
  int b = batch[n];
  const float* x = X + (size_t)n * MIN_DIM;
  atomicAdd(&sumX[b * MIN_DIM + k],            x[k]);
  atomicAdd(&sumX[b * MIN_DIM + KDIM + k],     x[KDIM + k]);
  atomicAdd(&sumX[b * MIN_DIM + 2 * KDIM + k], x[2 * KDIM + k]);
  if (k == 0) atomicAdd(&cnt[b], 1.f);
}

__global__ void centernorm_kernel(const int* __restrict__ batch, const float* __restrict__ X,
                                  const float* __restrict__ sumX, const float* __restrict__ cnt,
                                  float* __restrict__ Xc, float* __restrict__ normsum, int N) {
  int i = blockIdx.x * blockDim.x + threadIdx.x;
  if (i >= N * KDIM) return;
  int n = i >> 7, k = i & (KDIM - 1);
  int b = batch[n];
  float inv = 1.f / fmaxf(cnt[b], 1.f);
  const float* x = X + (size_t)n * MIN_DIM;
  float x0 = x[k]            - sumX[b * MIN_DIM + k] * inv;
  float x1 = x[KDIM + k]     - sumX[b * MIN_DIM + KDIM + k] * inv;
  float x2 = x[2 * KDIM + k] - sumX[b * MIN_DIM + 2 * KDIM + k] * inv;
  float* xc = Xc + (size_t)n * MIN_DIM;
  xc[k] = x0; xc[KDIM + k] = x1; xc[2 * KDIM + k] = x2;
  atomicAdd(&normsum[b * KDIM + k], sqrtf(x0 * x0 + x1 * x1 + x2 * x2));
}

__global__ void finalize_kernel(const int* __restrict__ batch, const float* __restrict__ cnt,
                                const float* __restrict__ normsum, const float* __restrict__ e3w,
                                float* __restrict__ Xn, float* __restrict__ out, int N) {
  int i = blockIdx.x * blockDim.x + threadIdx.x;
  if (i >= N * KDIM) return;
  int n = i >> 7, k = i & (KDIM - 1);
  int b = batch[n];
  float mn = normsum[b * KDIM + k] / fmaxf(cnt[b], 1.f);
  float scale = e3w[k] / (mn + 1e-5f);
  float* xn = Xn + (size_t)n * MIN_DIM;
  float* o  = out + (size_t)n * MIN_DIM;
#pragma unroll
  for (int d = 0; d < 3; ++d) {
    float v = xn[d * KDIM + k] * scale;
    xn[d * KDIM + k] = v;
    o[d * KDIM + k]  = v;
  }
}

__global__ __launch_bounds__(256) void layernorm_kernel(const float* __restrict__ H,
                                                        const float* __restrict__ gamma,
                                                        const float* __restrict__ beta,
                                                        float* __restrict__ Hn, int N) {
  int wave = threadIdx.x >> 5, lane = threadIdx.x & 31;
  int node = blockIdx.x * 8 + wave;
  if (node >= N) return;
  const float* h = H + (size_t)node * FDIM;
  float v0 = h[lane], v1 = h[lane + 32];
  float s = v0 + v1;
#pragma unroll
  for (int off = 16; off; off >>= 1) s += __shfl_xor(s, off, 32);
  float mu = s * (1.f / FDIM);
  float d0 = v0 - mu, d1 = v1 - mu;
  float q = d0 * d0 + d1 * d1;
#pragma unroll
  for (int off = 16; off; off >>= 1) q += __shfl_xor(q, off, 32);
  float inv = rsqrtf(q * (1.f / FDIM) + 1e-5f);
  float* o = Hn + (size_t)node * FDIM;
  o[lane]      = gamma[lane]      * d0 * inv + beta[lane];
  o[lane + 32] = gamma[lane + 32] * d1 * inv + beta[lane + 32];
}

// ---------------- fused edge-MLP + scatter kernel ----------------
// 8 waves/block, 16 edges/wave.  All WMMA under full EXEC; no workgroup
// barriers (same-wave LDS ordering only).
__global__ __launch_bounds__(256) void edge_kernel(
    const int* __restrict__ ei, const float* __restrict__ ea,
    const float* __restrict__ te, const float* __restrict__ Xn,
    const float* __restrict__ Hn, const bf16_t* __restrict__ Wf,
    const float* __restrict__ bm1, const float* __restrict__ bm2,
    const float* __restrict__ bx1, const float* __restrict__ bx2,
    float* __restrict__ out, int Ecnt) {
  __shared__ bf16_t sTile[8][16][FDIM];   // 16KB: act staging (D->A relayout)
  __shared__ float  sW[8][16][FDIM];      // 32KB: w staging (half of [16,128])
  __shared__ int    sSrc[8][16];
  __shared__ int    sTgt[8][16];

  const int wave = threadIdx.x >> 5;
  const int lane = threadIdx.x & 31;
  const int half = lane >> 4;
  const int col  = lane & 15;
  const int tileBase = (blockIdx.x * 8 + wave) * 16;

  if (lane < 16) {
    int e = tileBase + lane;
    int ec = e < Ecnt ? e : Ecnt - 1;
    sSrc[wave][lane] = ei[ec];          // edge_index[0] = src
    sTgt[wave][lane] = ei[Ecnt + ec];   // edge_index[1] = tgt
  }
  // same-wave DS ordering: reads below see the stores above

  const int m = col;                     // edge row this lane owns
  int eMine = tileBase + m;
  int ecMine = eMine < Ecnt ? eMine : Ecnt - 1;
  const int src = sSrc[wave][m];
  const int tgt = sTgt[wave][m];
  const float* Xs = Xn + (size_t)src * MIN_DIM;
  const float* Xt = Xn + (size_t)tgt * MIN_DIM;
  const float* Hs = Hn + (size_t)src * FDIM;
  const float* Ht = Hn + (size_t)tgt * FDIM;
  const float* EA = ea + (size_t)ecMine * FDIM;
  const float* TE = te + (size_t)ecMine * FDIM;

  // 8 consecutive m_in columns starting at base + half*8.  Each 16-col block
  // [base, base+16) lies in one source region, so the branch folds per base.
  // m_in: [H_tgt 0:64 | H_src 64:128 | edge_attr 128:192 | rel_dist 192:320 | te 320:384]
  auto gather8 = [&](int base) -> v8bf {
    int c0 = base + half * 8;
    if (base < 64) {
      const float4* p = (const float4*)(Ht + c0);
      return cvt8(p[0], p[1]);
    } else if (base < 128) {
      const float4* p = (const float4*)(Hs + (c0 - 64));
      return cvt8(p[0], p[1]);
    } else if (base < 192) {
      const float4* p = (const float4*)(EA + (c0 - 128));
      return cvt8(p[0], p[1]);
    } else if (base < 320) {
      int k0 = c0 - 192;
      float acc8[8] = {0.f, 0.f, 0.f, 0.f, 0.f, 0.f, 0.f, 0.f};
#pragma unroll
      for (int d = 0; d < 3; ++d) {
        float4 a0 = *(const float4*)(Xs + d * KDIM + k0);
        float4 a1 = *(const float4*)(Xs + d * KDIM + k0 + 4);
        float4 b0 = *(const float4*)(Xt + d * KDIM + k0);
        float4 b1 = *(const float4*)(Xt + d * KDIM + k0 + 4);
        float t0 = a0.x - b0.x; acc8[0] += t0 * t0;
        float t1 = a0.y - b0.y; acc8[1] += t1 * t1;
        float t2 = a0.z - b0.z; acc8[2] += t2 * t2;
        float t3 = a0.w - b0.w; acc8[3] += t3 * t3;
        float t4 = a1.x - b1.x; acc8[4] += t4 * t4;
        float t5 = a1.y - b1.y; acc8[5] += t5 * t5;
        float t6 = a1.z - b1.z; acc8[6] += t6 * t6;
        float t7 = a1.w - b1.w; acc8[7] += t7 * t7;
      }
      v8bf r;
#pragma unroll
      for (int q = 0; q < 8; ++q) r[q] = (bf16_t)acc8[q];
      return r;
    } else {
      const float4* p = (const float4*)(TE + (c0 - 320));
      return cvt8(p[0], p[1]);
    }
  };

  auto loadA_lds = [&](int kb) -> v16bf {
    const bf16_t* row = &sTile[wave][m][0];
    v8bf lo = *(const v8bf*)(row + kb + half * 8);
    v8bf hi = *(const v8bf*)(row + kb + 16 + half * 8);
    return __builtin_shufflevector(lo, hi, 0, 1, 2, 3, 4, 5, 6, 7, 8, 9, 10, 11,
                                   12, 13, 14, 15);
  };

  // ---- GEMM1: m_in[16,384] x Wm1[384,64], bias+SiLU -> sTile ----
  v8f acc[4];
#pragma unroll
  for (int t = 0; t < 4; ++t)
#pragma unroll
    for (int r = 0; r < 8; ++r) acc[t][r] = 0.f;

#pragma unroll
  for (int kb = 0; kb < 12; ++kb) {
    v16bf a = __builtin_shufflevector(gather8(kb * 32), gather8(kb * 32 + 16), 0, 1,
                                      2, 3, 4, 5, 6, 7, 8, 9, 10, 11, 12, 13, 14, 15);
#pragma unroll
    for (int t = 0; t < 4; ++t)
      acc[t] = wmma_bf16(a, load_frag(Wf, kb * 4 + t, lane), acc[t]);
  }
#pragma unroll
  for (int t = 0; t < 4; ++t) {
    int n = t * 16 + col;
    float bias = bm1[n];
#pragma unroll
    for (int r = 0; r < 8; ++r) {
      float p = acc[t][r] + bias;
      sTile[wave][r + half * 8][n] = (bf16_t)(p / (1.f + __expf(-p)));
    }
  }

  // ---- GEMM2: t1[16,64] x Wm2[64,64] + bm2 -> sTile ----
  {
    v8f a2[4];
#pragma unroll
    for (int t = 0; t < 4; ++t)
#pragma unroll
      for (int r = 0; r < 8; ++r) a2[t][r] = 0.f;
    v16bf aA = loadA_lds(0), aB = loadA_lds(32);
#pragma unroll
    for (int t = 0; t < 4; ++t) {
      a2[t] = wmma_bf16(aA, load_frag(Wf, 48 + t, lane), a2[t]);
      a2[t] = wmma_bf16(aB, load_frag(Wf, 52 + t, lane), a2[t]);
    }
#pragma unroll
    for (int t = 0; t < 4; ++t) {
      int n = t * 16 + col;
      float bias = bm2[n];
#pragma unroll
      for (int r = 0; r < 8; ++r)
        sTile[wave][r + half * 8][n] = (bf16_t)(a2[t][r] + bias);
    }
  }

  // ---- GEMM3: m_ij[16,64] x Wx1[64,64], bias+SiLU -> sTile ----
  {
    v8f a3[4];
#pragma unroll
    for (int t = 0; t < 4; ++t)
#pragma unroll
      for (int r = 0; r < 8; ++r) a3[t][r] = 0.f;
    v16bf aA = loadA_lds(0), aB = loadA_lds(32);
#pragma unroll
    for (int t = 0; t < 4; ++t) {
      a3[t] = wmma_bf16(aA, load_frag(Wf, 56 + t, lane), a3[t]);
      a3[t] = wmma_bf16(aB, load_frag(Wf, 60 + t, lane), a3[t]);
    }
#pragma unroll
    for (int t = 0; t < 4; ++t) {
      int n = t * 16 + col;
      float bias = bx1[n];
#pragma unroll
      for (int r = 0; r < 8; ++r) {
        float p = a3[t][r] + bias;
        sTile[wave][r + half * 8][n] = (bf16_t)(p / (1.f + __expf(-p)));
      }
    }
  }

  // ---- GEMM4: t2[16,64] x Wx2[64,128] -> w[16,128] in accumulators ----
  v8f w4[8];
#pragma unroll
  for (int t = 0; t < 8; ++t)
#pragma unroll
    for (int r = 0; r < 8; ++r) w4[t][r] = 0.f;
  {
    v16bf aA = loadA_lds(0), aB = loadA_lds(32);
#pragma unroll
    for (int t = 0; t < 8; ++t) {
      w4[t] = wmma_bf16(aA, load_frag(Wf, 64 + t, lane), w4[t]);
      w4[t] = wmma_bf16(aB, load_frag(Wf, 72 + t, lane), w4[t]);
    }
  }

  // ---- scatter in two k-halves: stage w (bias+clamp) through LDS so each
  //      lane owns edge=col with a contiguous 32-wide k chunk (float4 loads) ----
  float* o = out + (size_t)tgt * MIN_DIM;
  const int kc = half * 32;
#pragma unroll
  for (int hb = 0; hb < 2; ++hb) {
#pragma unroll
    for (int tt = 0; tt < 4; ++tt) {
      int t = hb * 4 + tt;
      int k = t * 16 + col;
      float bias = bx2[k];
#pragma unroll
      for (int r = 0; r < 8; ++r) {
        float wv = w4[t][r] + bias;
        wv = fminf(10.f, fmaxf(-10.f, wv));
        sW[wave][r + half * 8][tt * 16 + col] = wv;
      }
    }
    // same-wave LDS ordering: consume sees the stores above, and the next
    // half's stores cannot pass these loads.
    if (eMine < Ecnt) {
#pragma unroll
      for (int q = 0; q < 8; ++q) {
        int kk = kc + q * 4;            // column within this half
        int k = hb * 64 + kk;           // absolute k
        float4 wq = *(const float4*)&sW[wave][m][kk];
        float4 a0 = *(const float4*)(Xs + k);
        float4 b0 = *(const float4*)(Xt + k);
        float4 a1 = *(const float4*)(Xs + KDIM + k);
        float4 b1 = *(const float4*)(Xt + KDIM + k);
        float4 a2 = *(const float4*)(Xs + 2 * KDIM + k);
        float4 b2 = *(const float4*)(Xt + 2 * KDIM + k);
        float r0, r1, r2, rd, sc;
#define SCAT1(WC, C0, C1, C2, KOFF)                                        \
        r0 = (C0); r1 = (C1); r2 = (C2);                                   \
        rd = r0 * r0 + r1 * r1 + r2 * r2;                                  \
        sc = (WC) / (1.f + sqrtf(rd + 1e-8f));                             \
        atomicAdd(o + (KOFF), r0 * sc);                                    \
        atomicAdd(o + KDIM + (KOFF), r1 * sc);                             \
        atomicAdd(o + 2 * KDIM + (KOFF), r2 * sc);
        SCAT1(wq.x, a0.x - b0.x, a1.x - b1.x, a2.x - b2.x, k + 0)
        SCAT1(wq.y, a0.y - b0.y, a1.y - b1.y, a2.y - b2.y, k + 1)
        SCAT1(wq.z, a0.z - b0.z, a1.z - b1.z, a2.z - b2.z, k + 2)
        SCAT1(wq.w, a0.w - b0.w, a1.w - b1.w, a2.w - b2.w, k + 3)
#undef SCAT1
      }
    }
  }
}

// ---------------- launch ----------------

extern "C" void kernel_launch(void* const* d_in, const int* in_sizes, int n_in,
                              void* d_out, int out_size, void* d_ws, size_t ws_size,
                              hipStream_t stream) {
  const int*   batch = (const int*)d_in[0];
  const float* X     = (const float*)d_in[1];
  const float* H     = (const float*)d_in[2];
  const int*   ei    = (const int*)d_in[3];
  const float* ea    = (const float*)d_in[4];
  const float* te    = (const float*)d_in[5];
  const float* Wm1   = (const float*)d_in[6];
  const float* bm1   = (const float*)d_in[7];
  const float* Wm2   = (const float*)d_in[8];
  const float* bm2   = (const float*)d_in[9];
  const float* Wx1   = (const float*)d_in[10];
  const float* bx1   = (const float*)d_in[11];
  const float* Wx2   = (const float*)d_in[12];
  const float* bx2   = (const float*)d_in[13];
  const float* lng   = (const float*)d_in[14];
  const float* lnb   = (const float*)d_in[15];
  const float* e3w   = (const float*)d_in[16];

  const int N = in_sizes[0];
  const int E = in_sizes[3] / 2;

  // workspace layout
  float* ws      = (float*)d_ws;
  float* sumX    = ws;                                  // 64*384
  float* cnt     = sumX + NGRAPH * MIN_DIM;             // 64
  float* normsum = cnt + NGRAPH;                        // 64*128
  float* Xn      = normsum + NGRAPH * KDIM;             // N*384
  float* Hn      = Xn + (size_t)N * MIN_DIM;            // N*64
  bf16_t* Wf     = (bf16_t*)(Hn + (size_t)N * FDIM);    // 80*512 bf16 (fragment-packed)

  const int accumElems = NGRAPH * MIN_DIM + NGRAPH + NGRAPH * KDIM;
  zero_kernel<<<(accumElems + 255) / 256, 256, 0, stream>>>(ws, accumElems);

  wpack_kernel<<<(NFRAG * 512 + 255) / 256, 256, 0, stream>>>(Wm1, Wm2, Wx1, Wx2, Wf);

  const int NK = N * KDIM;
  accumx_kernel<<<(NK + 255) / 256, 256, 0, stream>>>(batch, X, sumX, cnt, N);
  centernorm_kernel<<<(NK + 255) / 256, 256, 0, stream>>>(batch, X, sumX, cnt, Xn, normsum, N);
  finalize_kernel<<<(NK + 255) / 256, 256, 0, stream>>>(batch, cnt, normsum, e3w, Xn,
                                                        (float*)d_out, N);
  layernorm_kernel<<<(N + 7) / 8, 256, 0, stream>>>(H, lng, lnb, Hn, N);

  const int tiles = (E + 15) / 16;
  const int blocks = (tiles + 7) / 8;
  edge_kernel<<<blocks, 256, 0, stream>>>(ei, ea, te, Xn, Hn, Wf, bm1, bm2, bx1, bx2,
                                          (float*)d_out, E);
}
